// CoOccurWithNorm_68865505624221
// MI455X (gfx1250) — compile-verified, matrix operand order checked by reading, not run on hardware
//
#include <hip/hip_runtime.h>
#include <hip/hip_bf16.h>

typedef __attribute__((ext_vector_type(16))) _Float16 v16h;
typedef __attribute__((ext_vector_type(8)))  _Float16 v8h;
typedef __attribute__((ext_vector_type(8)))  float    v8f;

#define NBINS   256
#define CHUNK   128                 // samples (K) staged in LDS per iteration
#define KSTEPS  (CHUNK / 32)        // 16x16x32 f16 WMMA steps per chunk
#define LDP     (CHUNK + 8)         // padded row stride (halves): 272B -> 4-bank step, conflict-free
#define LDS_HALFS (2 * NBINS * LDP) // A + B one-hot matrices
#define LDS_BYTES (LDS_HALFS * 2)

// ---------------------------------------------------------------------------
// Kernel 1: per-(image,channel) soft co-occurrence histogram as a one-hot GEMM
//   C[256x256] = Wx^T (256xN one-hot, f16) * Wy (Nx256 one-hot, f16), f32 accum
// One workgroup (512 threads / 16 waves) per slice; each wave owns a 4x4 block
// of 16x16 tiles -> 16 v_wmma_f32_16x16x32_f16 per K-step.
// ---------------------------------------------------------------------------
__global__ __launch_bounds__(512)
void hist_wmma_kernel(const float* __restrict__ X, float* __restrict__ out, int ntot)
{
    extern __shared__ _Float16 smem[];
    _Float16* Ah = smem;                 // [NBINS][LDP]  x one-hot (bin-major, K inner)
    _Float16* Bh = smem + NBINS * LDP;   // [NBINS][LDP]  y one-hot

    const int tid  = threadIdx.x;
    const int wave = tid >> 5;
    const int lane = tid & 31;
    const int half = lane >> 4;          // 0: K 0..7 / 16..23, 1: K 8..15 / 24..31
    const int mrow = lane & 15;          // A row / B column within tile
    const int brow = (wave >> 2) * 4;    // tile-row base of this wave's 4x4 block
    const int bcol = (wave & 3) * 4;     // tile-col base

    const float* Xp = X + (size_t)blockIdx.x * 512 * 512;

    // Zero one-hot LDS once; each chunk restores zeros after use.
    {
        uint4 z; z.x = 0u; z.y = 0u; z.z = 0u; z.w = 0u;
        uint4* p = (uint4*)smem;
        for (int i = tid; i < LDS_BYTES / 16; i += 512) p[i] = z;
    }

    const v8f vzero = {0.f, 0.f, 0.f, 0.f, 0.f, 0.f, 0.f, 0.f};
    v8f acc[4][4];
#pragma unroll
    for (int i = 0; i < 4; ++i)
#pragma unroll
        for (int j = 0; j < 4; ++j) acc[i][j] = vzero;

    const int chunks = (ntot + CHUNK - 1) / CHUNK;
    const float PI = 3.14159265358979f;

    for (int ch = 0; ch < chunks; ++ch) {
        // ---- scatter phase: 128 threads place 4 f16 weights each ----------
        int aix0 = 0, aix1 = 0, aiy0 = 0, aiy1 = 0;
        bool active = false;
        if (tid < CHUNK) {
            const int k = tid;
            const int n = ch * CHUNK + k;
            if (n < ntot) {
                active = true;
                const unsigned r = (unsigned)n / 511u;
                const unsigned j = (unsigned)n - r * 511u;
                const float x = Xp[r * 512u + j];
                const float y = Xp[r * 512u + j + 1u];

                // prefetch a future chunk's pixels (global_prefetch_b8)
                const int n2 = n + 4 * CHUNK;
                if (n2 < ntot) {
                    const unsigned r2 = (unsigned)n2 / 511u;
                    const unsigned j2 = (unsigned)n2 - r2 * 511u;
                    __builtin_prefetch(&Xp[r2 * 512u + j2], 0, 0);
                }

                // raised-cosine soft weights
                const float vx  = fminf(fmaxf(x, 0.f), 255.f);
                const float fx0 = floorf(vx);
                const float wx0 = 0.5f * (1.f + __cosf(PI * (vx - fx0)));
                const int   ix0 = (int)fx0;
                const int   ix1 = (ix0 + 1 > 255) ? 255 : ix0 + 1;

                const float vy  = fminf(fmaxf(y, 0.f), 255.f);
                const float fy0 = floorf(vy);
                const float wy0 = 0.5f * (1.f + __cosf(PI * (vy - fy0)));
                const int   iy0 = (int)fy0;
                const int   iy1 = (iy0 + 1 > 255) ? 255 : iy0 + 1;

                aix0 = ix0 * LDP + k; aix1 = ix1 * LDP + k;
                aiy0 = iy0 * LDP + k; aiy1 = iy1 * LDP + k;
                // i1 first, then i0: handles ix1==ix0 clip collision (w0 wins)
                Ah[aix1] = (_Float16)(1.f - wx0);
                Ah[aix0] = (_Float16)wx0;
                Bh[aiy1] = (_Float16)(1.f - wy0);
                Bh[aiy0] = (_Float16)wy0;
            }
        }
        __syncthreads();

        // ---- WMMA phase (uniform control flow, EXEC all-ones) -------------
#pragma unroll
        for (int ks = 0; ks < KSTEPS; ++ks) {
            const int kb = ks * 32 + half * 8;   // first K of this lane's low group
            v16h afrag[4];
#pragma unroll
            for (int i = 0; i < 4; ++i) {
                const _Float16* p = Ah + ((brow + i) * 16 + mrow) * LDP + kb;
                const v8h lo = *(const v8h*)p;         // K = kb .. kb+7      (elems 0..7)
                const v8h hi = *(const v8h*)(p + 16);  // K = kb+16 .. kb+23  (elems 8..15)
                afrag[i] = __builtin_shufflevector(lo, hi,
                    0, 1, 2, 3, 4, 5, 6, 7, 8, 9, 10, 11, 12, 13, 14, 15);
            }
#pragma unroll
            for (int jj = 0; jj < 4; ++jj) {
                const _Float16* p = Bh + ((bcol + jj) * 16 + mrow) * LDP + kb;
                const v8h lo = *(const v8h*)p;
                const v8h hi = *(const v8h*)(p + 16);
                const v16h bfrag = __builtin_shufflevector(lo, hi,
                    0, 1, 2, 3, 4, 5, 6, 7, 8, 9, 10, 11, 12, 13, 14, 15);
#pragma unroll
                for (int i = 0; i < 4; ++i) {
                    acc[i][jj] = __builtin_amdgcn_wmma_f32_16x16x32_f16(
                        false, afrag[i], false, bfrag,
                        (short)0, acc[i][jj], false, false);
                }
            }
        }
        __syncthreads();

        // ---- un-scatter: restore zeros at exactly the touched addresses ---
        if (active) {
            Ah[aix0] = (_Float16)0.f; Ah[aix1] = (_Float16)0.f;
            Bh[aiy0] = (_Float16)0.f; Bh[aiy1] = (_Float16)0.f;
        }
        // next chunk's scatter writes disjoint (column-owned) addresses;
        // visibility is enforced by the next iteration's first barrier.
    }

    // ---- write unnormalized histogram -------------------------------------
    // C/D 16x16 f32 layout: lane (half,mrow), VGPR v -> M = v + 8*half, N = mrow
    float* outp = out + (size_t)blockIdx.x * NBINS * NBINS;
#pragma unroll
    for (int i = 0; i < 4; ++i) {
        const int xbb = (brow + i) * 16 + half * 8;
#pragma unroll
        for (int jj = 0; jj < 4; ++jj) {
            const int yb = (bcol + jj) * 16 + mrow;
#pragma unroll
            for (int v = 0; v < 8; ++v)
                outp[(size_t)(xbb + v) * NBINS + yb] = acc[i][jj][v];
        }
    }
}

// ---------------------------------------------------------------------------
// Kernel 2: per-slice max reduction + in-place normalize (memory-trivial)
// ---------------------------------------------------------------------------
__global__ __launch_bounds__(256)
void norm_kernel(float* __restrict__ out)
{
    __shared__ float red[256];
    const int tid = threadIdx.x;
    float* p = out + (size_t)blockIdx.x * NBINS * NBINS;

    float m = 0.f;
    for (int i = tid; i < NBINS * NBINS; i += 256) m = fmaxf(m, p[i]);
    red[tid] = m;
    __syncthreads();
    for (int s = 128; s > 0; s >>= 1) {
        if (tid < s) red[tid] = fmaxf(red[tid], red[tid + s]);
        __syncthreads();
    }
    const float scale = 1.0f / red[0];
    for (int i = tid; i < NBINS * NBINS; i += 256) p[i] *= scale;
}

extern "C" void kernel_launch(void* const* d_in, const int* in_sizes, int n_in,
                              void* d_out, int out_size, void* d_ws, size_t ws_size,
                              hipStream_t stream)
{
    (void)n_in; (void)out_size; (void)d_ws; (void)ws_size;
    const float* X  = (const float*)d_in[0];
    float*       out = (float*)d_out;

    const int HW   = 512 * 512;
    const int BC   = in_sizes[0] / HW;   // 64 * 3 = 192 slices
    const int ntot = 511 * 512;          // adjacent pairs per slice

    hist_wmma_kernel<<<dim3(BC), dim3(512), LDS_BYTES, stream>>>(X, out, ntot);
    norm_kernel<<<dim3(BC), dim3(256), 0, stream>>>(out);
}
